// Module_GAT_VC_split_49254684950642
// MI455X (gfx1250) — matrix operand order repaired and checked by reading, not compiled, not run on hardware
//
#include <hip/hip_runtime.h>
#include <hip/hip_bf16.h>

// ---------------------------------------------------------------------------
// Types / constants
// ---------------------------------------------------------------------------
typedef __bf16 bf16_t;
typedef bf16_t v16bf __attribute__((ext_vector_type(16)));
typedef float  v8f   __attribute__((ext_vector_type(8)));

#define BATCH 4
#define SEQ   512
#define BN    (BATCH * SEQ)   // 2048 rows total
#define DIN   64
#define DATT  32
#define DHID  128
#define AHID  64
#define MASKC 10000.0f

// ---------------------------------------------------------------------------
// WMMA helpers (CDNA5 wave32, v_wmma_f32_16x16x32_bf16)
// ---------------------------------------------------------------------------
__device__ __forceinline__ v8f wmma_bf16(v16bf a, v16bf b, v8f c) {
  // (neg_a, A, neg_b, B, c_mod, C, reuse_a, reuse_b)
  return __builtin_amdgcn_wmma_f32_16x16x32_bf16(false, a, false, b,
                                                 (short)0, c, false, false);
}

// A-matrix 16x32 bf16 fragment from row-major f32 [16 x ld].
// Layout (ISA 7.12.2): lanes 0-15 row M=lane, K = 0..7 & 16..23;
//                      lanes 16-31 row M=lane-16, K = 8..15 & 24..31.
__device__ __forceinline__ v16bf ld_a_f32(const float* src, int ld, int lane, int k0) {
  const int r  = lane & 15;
  const int kb = k0 + ((lane >> 4) << 3);
  const float* p = src + r * ld + kb;
  v16bf a;
#pragma unroll
  for (int u = 0; u < 8; ++u) a[u] = (bf16_t)p[u];
#pragma unroll
  for (int u = 0; u < 8; ++u) a[8 + u] = (bf16_t)p[16 + u];
  return a;
}

// Same, but source is bf16 row-major in LDS.
__device__ __forceinline__ v16bf ld_a_ldsbf(const bf16_t* src, int ld, int lane, int k0) {
  const int r  = lane & 15;
  const int kb = k0 + ((lane >> 4) << 3);
  const bf16_t* p = src + r * ld + kb;
  v16bf a;
#pragma unroll
  for (int u = 0; u < 8; ++u) a[u] = p[u];
#pragma unroll
  for (int u = 0; u < 8; ++u) a[8 + u] = p[16 + u];
  return a;
}

// B-matrix 32x16 bf16 fragment from TRANSPOSED weights wT[N][K] row-major.
// Layout: lanes 0-15 column N=lane, K=0..7 & 16..23; lanes 16-31 column N=lane-16, K=8..15 & 24..31.
// -> per lane two runs of 8 bf16 (16 bytes each): two global_load_b128.
__device__ __forceinline__ v16bf ld_b_T(const bf16_t* wT, int ldK, int lane, int n0, int k0) {
  const bf16_t* p = wT + (size_t)(n0 + (lane & 15)) * ldK + k0 + ((lane >> 4) << 3);
  v16bf b;
#pragma unroll
  for (int u = 0; u < 8; ++u) b[u] = p[u];
#pragma unroll
  for (int u = 0; u < 8; ++u) b[8 + u] = p[16 + u];
  return b;
}

// ---------------------------------------------------------------------------
// CDNA5 async global->LDS (ASYNCcnt path, ISA 10 / 15.18.3)
// ---------------------------------------------------------------------------
__device__ __forceinline__ unsigned lds_off(const void* p) {
  // generic (LDS-resident) pointer -> LDS byte offset
  return (unsigned)(unsigned long long)(const __attribute__((address_space(3))) void*)p;
}

__device__ __forceinline__ void async_ld_b128(unsigned lds, const void* gsrc) {
  asm volatile("global_load_async_to_lds_b128 %0, %1, off"
               :: "v"(lds), "v"((unsigned long long)gsrc)
               : "memory");
}

__device__ __forceinline__ void wait_async0() {
#if __has_builtin(__builtin_amdgcn_s_wait_asynccnt)
  __builtin_amdgcn_s_wait_asynccnt(0);
#else
  asm volatile("s_wait_asynccnt 0" ::: "memory");
#endif
}

// stage one 32x64 f32 hq block (8KB): 128 threads x 4 b128 DMA issues
__device__ __forceinline__ void stage_hq_async(float* dstLds, const float* src, int tid) {
#pragma unroll
  for (int u = 0; u < 4; ++u) {
    const int off = tid * 4 + u * 512;  // float units, 16B aligned
    async_ld_b128(lds_off(dstLds + off), (const void*)(src + off));
  }
}

// ---------------------------------------------------------------------------
// Kernel 0: f32 [K x N] weight -> bf16 transposed [N x K]
// ---------------------------------------------------------------------------
__global__ void k_transpose_bf16(const float* __restrict__ src, bf16_t* __restrict__ dst,
                                 int K, int N) {
  int idx = blockIdx.x * 256 + threadIdx.x;
  if (idx < K * N) {
    int k = idx / N, n = idx - k * N;
    dst[(size_t)n * K + k] = (bf16_t)src[idx];
  }
}

// ---------------------------------------------------------------------------
// Kernel 1: encoders. Per wave: one 16-row tile of [BN].
// ---------------------------------------------------------------------------
__global__ __launch_bounds__(256) void k_encode(
    const float* __restrict__ x,
    const float* __restrict__ kb1, const float* __restrict__ kb2,
    const float* __restrict__ qb1, const float* __restrict__ qb2,
    const float* __restrict__ vb1, const float* __restrict__ vb2,
    const float* __restrict__ ab1,
    const bf16_t* __restrict__ kW1T, const bf16_t* __restrict__ kW2T,
    const bf16_t* __restrict__ qW1T, const bf16_t* __restrict__ qW2T,
    const bf16_t* __restrict__ vW1T, const bf16_t* __restrict__ vW2T,
    const bf16_t* __restrict__ aW1kT, const bf16_t* __restrict__ aW1qT,
    float* __restrict__ hk, float* __restrict__ hq,
    float* __restrict__ v_f32, bf16_t* __restrict__ vT) {
  __shared__ bf16_t s_h1[8][16 * DHID];  // 32 KB
  __shared__ bf16_t s_k [8][16 * DATT];  // 8 KB
  __shared__ bf16_t s_q [8][16 * DATT];  // 8 KB

  const int lane = threadIdx.x & 31;
  const int wave = threadIdx.x >> 5;
  const int tile = blockIdx.x * 8 + wave;
  const int R0   = tile * 16;
  const int col  = lane & 15;
  const int mb   = (lane >> 4) << 3;  // row base of this lane's acc half

  const v16bf ax0 = ld_a_f32(x + (size_t)R0 * DIN, DIN, lane, 0);
  const v16bf ax1 = ld_a_f32(x + (size_t)R0 * DIN, DIN, lane, 32);

  const bf16_t* W1T[3] = {kW1T, qW1T, vW1T};
  const bf16_t* W2T[3] = {kW2T, qW2T, vW2T};
  const float*  B1 [3] = {kb1, qb1, vb1};
  const float*  B2 [3] = {kb2, qb2, vb2};

  bf16_t* h1 = s_h1[wave];

#pragma unroll
  for (int e = 0; e < 3; ++e) {
    // ---- layer 1: [16x64] @ [64x128] + b1, relu -> LDS bf16 ----
#pragma unroll
    for (int nt = 0; nt < 8; ++nt) {
      const int n0 = nt * 16;
      const float bb = B1[e][n0 + col];
      v8f acc;
#pragma unroll
      for (int r = 0; r < 8; ++r) acc[r] = bb;
      acc = wmma_bf16(ax0, ld_b_T(W1T[e], DIN, lane, n0, 0),  acc);
      acc = wmma_bf16(ax1, ld_b_T(W1T[e], DIN, lane, n0, 32), acc);
#pragma unroll
      for (int r = 0; r < 8; ++r) {
        const int m = r + mb;
        h1[m * DHID + n0 + col] = (bf16_t)fmaxf(acc[r], 0.0f);
      }
    }
    // ---- layer 2: [16x128] @ [128x32] + b2 ----
#pragma unroll
    for (int nt = 0; nt < 2; ++nt) {
      const int n0 = nt * 16;
      const float bb = B2[e][n0 + col];
      v8f acc;
#pragma unroll
      for (int r = 0; r < 8; ++r) acc[r] = bb;
#pragma unroll
      for (int ks = 0; ks < 4; ++ks)
        acc = wmma_bf16(ld_a_ldsbf(h1, DHID, lane, ks * 32),
                        ld_b_T(W2T[e], DHID, lane, n0, ks * 32), acc);
      if (e == 0) {
#pragma unroll
        for (int r = 0; r < 8; ++r) s_k[wave][(r + mb) * DATT + n0 + col] = (bf16_t)acc[r];
      } else if (e == 1) {
#pragma unroll
        for (int r = 0; r < 8; ++r) s_q[wave][(r + mb) * DATT + n0 + col] = (bf16_t)acc[r];
      } else {  // v -> f32 row-major + bf16 transposed per batch
        const int bidx = R0 / SEQ;
        const int jj0  = R0 - bidx * SEQ;
#pragma unroll
        for (int r = 0; r < 8; ++r) {
          const int m = r + mb;
          v_f32[(size_t)(R0 + m) * DATT + n0 + col] = acc[r];
          vT[((size_t)bidx * DATT + n0 + col) * SEQ + jj0 + m] = (bf16_t)acc[r];
        }
      }
    }
  }

  // ---- hk = k @ aW1[:32] + ab1 ; hq = q @ aW1[32:] : [16x32]@[32x64] ----
#pragma unroll
  for (int nt = 0; nt < 4; ++nt) {
    const int n0 = nt * 16;
    const float bb = ab1[n0 + col];
    v8f acck, accq;
#pragma unroll
    for (int r = 0; r < 8; ++r) { acck[r] = bb; accq[r] = 0.0f; }
    acck = wmma_bf16(ld_a_ldsbf(s_k[wave], DATT, lane, 0),
                     ld_b_T(aW1kT, DATT, lane, n0, 0), acck);
    accq = wmma_bf16(ld_a_ldsbf(s_q[wave], DATT, lane, 0),
                     ld_b_T(aW1qT, DATT, lane, n0, 0), accq);
#pragma unroll
    for (int r = 0; r < 8; ++r) {
      const int m = r + mb;
      hk[(size_t)(R0 + m) * AHID + n0 + col] = acck[r];
      hq[(size_t)(R0 + m) * AHID + n0 + col] = accq[r];
    }
  }
}

// ---------------------------------------------------------------------------
// Kernel 2: pairwise attention + aggregation, with double-buffered
// global_load_async_to_lds_b128 staging of hq j-blocks (ASYNCcnt path).
// Block = 128 thr (4 waves): (batch b, 64 i-rows). Each wave: 16 i-rows.
// ---------------------------------------------------------------------------
__global__ __launch_bounds__(128) void k_pair(
    const float* __restrict__ hk, const float* __restrict__ hq,
    const float* __restrict__ aW2, const float* __restrict__ ab2,
    const bf16_t* __restrict__ vT,
    float* __restrict__ agg0, float* __restrict__ agg1) {
  __shared__ float s_hk[64 * AHID];      // 16 KB: this block's hk rows
  __shared__ float s_hq[2][32 * AHID];   // 2 x 8 KB: double-buffered j-blocks
  __shared__ float s_aw2[2 * AHID];      // 512 B

  const int lane = threadIdx.x & 31;
  const int wave = threadIdx.x >> 5;
  const int b    = blockIdx.y;
  const int iblk = blockIdx.x;                           // 64-row i-block within batch
  const int irow = iblk * 64 + wave * 16 + (lane & 15);  // this lane's A row (in batch)

  const float* hqb = hq + (size_t)b * SEQ * AHID;

  // kick off async DMA of the first hq block while we stage hk/aW2
  stage_hq_async(s_hq[0], hqb, threadIdx.x);

  s_aw2[threadIdx.x] = aW2[threadIdx.x];  // 128 threads == 128 elems
  const float* hkb = hk + ((size_t)b * SEQ + iblk * 64) * AHID;
#pragma unroll
  for (int u = 0; u < 32; ++u) {
    const int idx = threadIdx.x + u * 128;
    s_hk[idx] = hkb[idx];
  }

  const float c0 = ab2[0], c1 = ab2[1];
  const float* hkrow = &s_hk[(wave * 16 + (lane & 15)) * AHID];
  const bf16_t* vTb = vT + (size_t)b * DATT * SEQ;

  v8f a00 = {}, a01 = {}, a10 = {}, a11 = {};  // {chan}{d-tile}
  const int kb = (lane >> 4) << 3;

  wait_async0();
  __syncthreads();  // first hq block + hk/aW2 visible to all waves

  for (int jb = 0; jb < SEQ / 32; ++jb) {
    const int j0 = jb * 32;
    const float* cur = s_hq[jb & 1];

    // overlap DMA of next block with this block's score + WMMA compute
    if (jb + 1 < SEQ / 32)
      stage_hq_async(s_hq[(jb + 1) & 1], hqb + (size_t)(j0 + 32) * AHID, threadIdx.x);

    // w fragments directly in A layout: elem t<8 -> K=kb+t ; t>=8 -> K=kb+8+t
    v16bf fa0, fa1;
#pragma unroll
    for (int t = 0; t < 16; ++t) {
      const int kk = kb + ((t < 8) ? t : (8 + t));
      const int j  = j0 + kk;
      const float* hqrow = &cur[kk * AHID];
      float z0 = 0.0f, z1 = 0.0f;
      for (int a = 0; a < AHID; ++a) {
        const float s = fmaxf(hkrow[a] + hqrow[a], 0.0f);
        z0 = fmaf(s, s_aw2[2 * a],     z0);
        z1 = fmaf(s, s_aw2[2 * a + 1], z1);
      }
      z0 += c0; z1 += c1;
      if (j == irow) { z0 -= MASKC; z1 -= MASKC; }
      fa0[t] = (bf16_t)(1.0f / (1.0f + __expf(-z0)));
      fa1[t] = (bf16_t)(1.0f / (1.0f + __expf(-z1)));
    }

    const v16bf fb0 = ld_b_T(vTb, SEQ, lane, 0,  j0);
    const v16bf fb1 = ld_b_T(vTb, SEQ, lane, 16, j0);
    a00 = wmma_bf16(fa0, fb0, a00);
    a01 = wmma_bf16(fa0, fb1, a01);
    a10 = wmma_bf16(fa1, fb0, a10);
    a11 = wmma_bf16(fa1, fb1, a11);

    if (jb + 1 < SEQ / 32) {
      wait_async0();     // this wave's DMA for block jb+1 done
      __syncthreads();   // all waves' DMA done; also fences buffer reuse
    }
  }

  const int col = lane & 15;
  const int mb  = (lane >> 4) << 3;
#pragma unroll
  for (int r = 0; r < 8; ++r) {
    const size_t g = (size_t)b * SEQ + iblk * 64 + wave * 16 + mb + r;
    agg0[g * DATT + col]      = a00[r];
    agg0[g * DATT + 16 + col] = a01[r];
    agg1[g * DATT + col]      = a10[r];
    agg1[g * DATT + 16 + col] = a11[r];
  }
}

// ---------------------------------------------------------------------------
// Kernel 3: decoder + heads. d = relu([v|agg]@W1+b1)@W2+b2; mu/softplus-sig.
// ---------------------------------------------------------------------------
__global__ __launch_bounds__(256) void k_decode(
    const float* __restrict__ v_f32, const float* __restrict__ agg,
    const float* __restrict__ b1, const float* __restrict__ b2,
    const bf16_t* __restrict__ W1T, const bf16_t* __restrict__ W2T,
    const float* __restrict__ mW, const float* __restrict__ mb_,
    const float* __restrict__ sW, const float* __restrict__ sb_,
    float* __restrict__ out_mu, float* __restrict__ out_sig) {
  __shared__ bf16_t s_h1[8][16 * DHID];  // 32 KB
  __shared__ float  s_d [8][16 * AHID];  // 32 KB

  const int lane = threadIdx.x & 31;
  const int wave = threadIdx.x >> 5;
  const int tile = blockIdx.x * 8 + wave;
  const int R0   = tile * 16;
  const int col  = lane & 15;
  const int mbr  = (lane >> 4) << 3;

  // c = [v | agg]: K 0..31 from v, K 32..63 from agg
  const v16bf ac0 = ld_a_f32(v_f32 + (size_t)R0 * DATT, DATT, lane, 0);
  const v16bf ac1 = ld_a_f32(agg   + (size_t)R0 * DATT, DATT, lane, 0);

  bf16_t* h1 = s_h1[wave];
#pragma unroll
  for (int nt = 0; nt < 8; ++nt) {
    const int n0 = nt * 16;
    const float bb = b1[n0 + col];
    v8f acc;
#pragma unroll
    for (int r = 0; r < 8; ++r) acc[r] = bb;
    acc = wmma_bf16(ac0, ld_b_T(W1T, 2 * DATT, lane, n0, 0),  acc);
    acc = wmma_bf16(ac1, ld_b_T(W1T, 2 * DATT, lane, n0, 32), acc);
#pragma unroll
    for (int r = 0; r < 8; ++r)
      h1[(r + mbr) * DHID + n0 + col] = (bf16_t)fmaxf(acc[r], 0.0f);
  }

  float* dd = s_d[wave];
#pragma unroll
  for (int nt = 0; nt < 4; ++nt) {
    const int n0 = nt * 16;
    const float bb = b2[n0 + col];
    v8f acc;
#pragma unroll
    for (int r = 0; r < 8; ++r) acc[r] = bb;
#pragma unroll
    for (int ks = 0; ks < 4; ++ks)
      acc = wmma_bf16(ld_a_ldsbf(h1, DHID, lane, ks * 32),
                      ld_b_T(W2T, DHID, lane, n0, ks * 32), acc);
#pragma unroll
    for (int r = 0; r < 8; ++r) dd[(r + mbr) * AHID + n0 + col] = acc[r];
  }

  // heads: lanes 0-15 -> mu row L; lanes 16-31 -> softplus-sig row L-16
  const int r = lane & 15;
  const float* drow = &dd[r * AHID];
  float z = 0.0f;
  if (lane < 16) {
    for (int a = 0; a < AHID; ++a) z = fmaf(drow[a], mW[a], z);
    out_mu[R0 + r] = z + mb_[0];
  } else {
    for (int a = 0; a < AHID; ++a) z = fmaf(drow[a], sW[a], z);
    z += sb_[0];
    out_sig[R0 + r] = (z > 0.0f) ? (z + log1pf(__expf(-z))) : log1pf(__expf(z));
  }
}

// ---------------------------------------------------------------------------
// Host launcher
// ---------------------------------------------------------------------------
extern "C" void kernel_launch(void* const* d_in, const int* in_sizes, int n_in,
                              void* d_out, int out_size, void* d_ws, size_t ws_size,
                              hipStream_t stream) {
  const float* x    = (const float*)d_in[0];
  const float* kW1  = (const float*)d_in[1];  const float* kb1 = (const float*)d_in[2];
  const float* kW2  = (const float*)d_in[3];  const float* kb2 = (const float*)d_in[4];
  const float* qW1  = (const float*)d_in[5];  const float* qb1 = (const float*)d_in[6];
  const float* qW2  = (const float*)d_in[7];  const float* qb2 = (const float*)d_in[8];
  const float* vW1  = (const float*)d_in[9];  const float* vb1 = (const float*)d_in[10];
  const float* vW2  = (const float*)d_in[11]; const float* vb2 = (const float*)d_in[12];
  const float* aW1  = (const float*)d_in[13]; const float* ab1 = (const float*)d_in[14];
  const float* aW2  = (const float*)d_in[15]; const float* ab2 = (const float*)d_in[16];
  const float* dxW1 = (const float*)d_in[17]; const float* dxb1= (const float*)d_in[18];
  const float* dxW2 = (const float*)d_in[19]; const float* dxb2= (const float*)d_in[20];
  const float* dyW1 = (const float*)d_in[21]; const float* dyb1= (const float*)d_in[22];
  const float* dyW2 = (const float*)d_in[23]; const float* dyb2= (const float*)d_in[24];
  const float* mxW  = (const float*)d_in[25]; const float* mxb = (const float*)d_in[26];
  const float* sxW  = (const float*)d_in[27]; const float* sxb = (const float*)d_in[28];
  const float* myW  = (const float*)d_in[29]; const float* myb = (const float*)d_in[30];
  const float* syW  = (const float*)d_in[31]; const float* syb = (const float*)d_in[32];
  float* out = (float*)d_out;

  char* ws = (char*)d_ws;
  size_t off = 0;
  auto alloc = [&](size_t bytes) -> void* {
    void* p = ws + off;
    off = (off + bytes + 255) & ~(size_t)255;
    return p;
  };
  bf16_t* kW1T  = (bf16_t*)alloc(DIN * DHID * 2);
  bf16_t* qW1T  = (bf16_t*)alloc(DIN * DHID * 2);
  bf16_t* vW1T  = (bf16_t*)alloc(DIN * DHID * 2);
  bf16_t* kW2T  = (bf16_t*)alloc(DHID * DATT * 2);
  bf16_t* qW2T  = (bf16_t*)alloc(DHID * DATT * 2);
  bf16_t* vW2T  = (bf16_t*)alloc(DHID * DATT * 2);
  bf16_t* aW1kT = (bf16_t*)alloc(DATT * AHID * 2);
  bf16_t* aW1qT = (bf16_t*)alloc(DATT * AHID * 2);
  bf16_t* dxW1T = (bf16_t*)alloc(2 * DATT * DHID * 2);
  bf16_t* dyW1T = (bf16_t*)alloc(2 * DATT * DHID * 2);
  bf16_t* dxW2T = (bf16_t*)alloc(DHID * AHID * 2);
  bf16_t* dyW2T = (bf16_t*)alloc(DHID * AHID * 2);
  float*  hk    = (float*)alloc((size_t)BN * AHID * 4);
  float*  hq    = (float*)alloc((size_t)BN * AHID * 4);
  float*  v_f32 = (float*)alloc((size_t)BN * DATT * 4);
  bf16_t* vT    = (bf16_t*)alloc((size_t)BATCH * DATT * SEQ * 2);
  float*  agg0  = (float*)alloc((size_t)BN * DATT * 4);
  float*  agg1  = (float*)alloc((size_t)BN * DATT * 4);
  (void)ws_size; (void)n_in; (void)in_sizes; (void)out_size;

  auto tr = [&](const float* s, bf16_t* d, int K, int N) {
    k_transpose_bf16<<<(K * N + 255) / 256, 256, 0, stream>>>(s, d, K, N);
  };
  tr(kW1, kW1T, DIN, DHID);   tr(qW1, qW1T, DIN, DHID);   tr(vW1, vW1T, DIN, DHID);
  tr(kW2, kW2T, DHID, DATT);  tr(qW2, qW2T, DHID, DATT);  tr(vW2, vW2T, DHID, DATT);
  tr(aW1, aW1kT, DATT, AHID); tr(aW1 + DATT * AHID, aW1qT, DATT, AHID);
  tr(dxW1, dxW1T, 2 * DATT, DHID); tr(dyW1, dyW1T, 2 * DATT, DHID);
  tr(dxW2, dxW2T, DHID, AHID);     tr(dyW2, dyW2T, DHID, AHID);

  // encoders: 2048 rows / 16 per wave / 8 waves = 16 blocks
  k_encode<<<16, 256, 0, stream>>>(x, kb1, kb2, qb1, qb2, vb1, vb2, ab1,
                                   kW1T, kW2T, qW1T, qW2T, vW1T, vW2T,
                                   aW1kT, aW1qT, hk, hq, v_f32, vT);

  // pairwise + aggregation: grid (8 i-blocks x 4 batches), 128 threads
  k_pair<<<dim3(8, BATCH), 128, 0, stream>>>(hk, hq, aW2, ab2, vT, agg0, agg1);

  // decoders + heads: out = [mu_x | sig_x | mu_y | sig_y], each BN floats
  k_decode<<<16, 256, 0, stream>>>(v_f32, agg0, dxb1, dxb2, dxW1T, dxW2T,
                                   mxW, mxb, sxW, sxb, out, out + BN);
  k_decode<<<16, 256, 0, stream>>>(v_f32, agg1, dyb1, dyb2, dyW1T, dyW2T,
                                   myW, myb, syW, syb, out + 2 * BN, out + 3 * BN);
}
